// MHA_79156247265387
// MI455X (gfx1250) — matrix operand order repaired and checked by reading, not compiled, not run on hardware
//
#include <hip/hip_runtime.h>

// ---------------------------------------------------------------------------
// MHA for MI455X (gfx1250): bf16 WMMA (v_wmma_f32_16x16x32_bf16) pipeline.
// ---------------------------------------------------------------------------

#define B_   8
#define H_   9
#define E_   128
#define S_   1025
#define HE_  (H_ * E_)      // 1152
#define SP_  1056           // padded S stride for vT (multiple of 32, 16B-aligned rows)

typedef __attribute__((ext_vector_type(16))) __bf16 v16bf;
typedef __attribute__((ext_vector_type(8)))  float  v8f;

typedef unsigned short u16;

union FragU { uint4 q[2]; v16bf v; };

__device__ __forceinline__ v8f zero_v8f() {
  v8f z = {0.f, 0.f, 0.f, 0.f, 0.f, 0.f, 0.f, 0.f};
  return z;
}

// f32 -> bf16 (round to nearest even), raw u16 bits
__device__ __forceinline__ u16 f2bf_u16(float f) {
  unsigned u = __float_as_uint(f);
  u += 0x7FFFu + ((u >> 16) & 1u);
  return (u16)(u >> 16);
}

__device__ __forceinline__ v8f wmma_bf16(v16bf a, v16bf b, v8f c) {
  // D = A(16x32 bf16) * B(32x16 bf16) + C(16x16 f32)
  return __builtin_amdgcn_wmma_f32_16x16x32_bf16(
      /*neg_a=*/false, a, /*neg_b=*/false, b,
      /*c_mod=*/(short)0, c, /*reuse_a=*/false, /*reuse_b=*/false);
}

// A fragment, 16x32 bf16 (ISA 7.12.2): lane l: m=l&15, g=l>>4
//   VGPR0-3: K = 8g..8g+7 ; VGPR4-7: K = 16+8g..16+8g+7
__device__ __forceinline__ v16bf frag_A(const u16* t /*16 rows x 32, ldk=32*/, int lane) {
  int m = lane & 15, g = lane >> 4;
  const u16* p = t + m * 32;
  FragU f;
  f.q[0] = *(const uint4*)(p + 8 * g);
  f.q[1] = *(const uint4*)(p + 16 + 8 * g);
  return f.v;
}

// B fragment from Bt[n][k] tile (B stored transposed in LDS).
// Dense-B lane layout (matches frag_A): lane l: n=l&15, g=l>>4,
// halves 0..7 hold K = 8g..8g+7, halves 8..15 hold K = 16+8g..16+8g+7.
__device__ __forceinline__ v16bf frag_B(const u16* t /*16 rows(n) x 32(k), ldk=32*/, int lane) {
  int n = lane & 15, g = lane >> 4;
  const u16* p = t + n * 32 + 8 * g;
  FragU f;
  f.q[0] = *(const uint4*)(p);
  f.q[1] = *(const uint4*)(p + 16);
  return f.v;
}

// Copy a [rows x 32] u16(bf16) tile from global (row-major, stride ld) into LDS,
// zero-padding out-of-range elements. 256 threads.
__device__ __forceinline__ void load_tile_u16(u16* __restrict__ dst,
                                              const u16* __restrict__ src,
                                              long ld, long r0, long c0,
                                              long rmax, long cmax,
                                              int rows, int tid) {
  int chunks = rows * 4;                       // 4 x 8-half chunks per row
  for (int ch = tid; ch < chunks; ch += 256) {
    int row = ch >> 2;
    int col = (ch & 3) << 3;
    long gr = r0 + row, gc = c0 + col;
    if (gr < rmax && gc + 7 < cmax) {
      *(uint4*)(dst + row * 32 + col) = *(const uint4*)(src + gr * ld + gc);
    } else {
#pragma unroll
      for (int i = 0; i < 8; i++) {
        u16 v = 0;
        if (gr < rmax && gc + i < cmax) v = src[gr * ld + gc + i];
        dst[row * 32 + col + i] = v;
      }
    }
  }
}

// Copy a [64 x 32] tile from an f32 matrix, converting to bf16 into LDS.
__device__ __forceinline__ void load_tile_f32_bf16(u16* __restrict__ dst,
                                                   const float* __restrict__ src,
                                                   long ld, long r0, long c0,
                                                   long rmax, long cmax, int tid) {
  for (int idx = tid; idx < 64 * 32; idx += 256) {
    int row = idx >> 5, col = idx & 31;
    long gr = r0 + row, gc = c0 + col;
    float v = 0.f;
    if (gr < rmax && gc < cmax) v = src[gr * ld + gc];
    dst[idx] = f2bf_u16(v);
  }
}

// ---------------------------------------------------------------------------
// Conversion kernels
// ---------------------------------------------------------------------------

// xt[b][s][e] = bf16(x[b][e][s])
__global__ void cvt_x_kernel(const float* __restrict__ x, u16* __restrict__ xt) {
  long i = (long)blockIdx.x * 256 + threadIdx.x;
  const long N = (long)B_ * S_ * E_;
  if (i >= N) return;
  int e = (int)(i % E_);
  long t = i / E_;
  int s = (int)(t % S_);
  int b = (int)(t / S_);
  xt[i] = f2bf_u16(x[((long)b * E_ + e) * S_ + s]);
}

// dst[h][f][e] = bf16(src[h][e][f])  (transpose each head matrix)
__global__ void cvt_headT_kernel(const float* __restrict__ src, u16* __restrict__ dst) {
  long i = (long)blockIdx.x * 256 + threadIdx.x;
  const long N = (long)H_ * E_ * E_;
  if (i >= N) return;
  int e = (int)(i % E_);
  long t = i / E_;
  int f = (int)(t % E_);
  int h = (int)(t / E_);
  dst[i] = f2bf_u16(src[((long)h * E_ + e) * E_ + f]);
}

__global__ void cvt_plain_kernel(const float* __restrict__ src, u16* __restrict__ dst, long n) {
  long i = (long)blockIdx.x * 256 + threadIdx.x;
  if (i < n) dst[i] = f2bf_u16(src[i]);
}

// ---------------------------------------------------------------------------
// GEMM1: per (b,h,mode): C[S,E] = Xt[b](S x E) * P[h](E x E)
//   mode 0 -> q (scale folded in), mode 1 -> k, mode 2 -> v stored transposed
// ---------------------------------------------------------------------------
__global__ __launch_bounds__(256) void gemm_proj_kernel(
    const u16* __restrict__ xt, const u16* __restrict__ QT,
    const u16* __restrict__ KT, const u16* __restrict__ VT,
    u16* __restrict__ qb, u16* __restrict__ kb, u16* __restrict__ vT) {
  __shared__ alignas(16) u16 As[64 * 32];
  __shared__ alignas(16) u16 Bs[128 * 32];
  int tid = threadIdx.x, lane = tid & 31, wave = tid >> 5;
  int wm0 = (wave >> 2) * 32, wn0 = (wave & 3) * 32;
  int z = blockIdx.z, mode = z % 3, bh = z / 3;
  int h = bh % H_, b = bh / H_;
  long m0 = (long)blockIdx.y * 64;
  const u16* A = xt + (long)b * S_ * E_;
  const u16* Bt = (mode == 0 ? QT : (mode == 1 ? KT : VT)) + (long)h * E_ * E_;

  v8f acc00 = zero_v8f(), acc01 = zero_v8f(), acc10 = zero_v8f(), acc11 = zero_v8f();
  for (int k0 = 0; k0 < E_; k0 += 32) {
    load_tile_u16(As, A, E_, m0, k0, S_, E_, 64, tid);
    load_tile_u16(Bs, Bt, E_, 0, k0, E_, E_, 128, tid);
    __syncthreads();
    v16bf a0 = frag_A(As + wm0 * 32, lane);
    v16bf a1 = frag_A(As + (wm0 + 16) * 32, lane);
    v16bf b0 = frag_B(Bs + wn0 * 32, lane);
    v16bf b1 = frag_B(Bs + (wn0 + 16) * 32, lane);
    acc00 = wmma_bf16(a0, b0, acc00);
    acc01 = wmma_bf16(a0, b1, acc01);
    acc10 = wmma_bf16(a1, b0, acc10);
    acc11 = wmma_bf16(a1, b1, acc11);
    __syncthreads();
  }
  const float scale = 0.08838834764831845f;  // 1/sqrt(128)
  int g = lane >> 4, nl = lane & 15;
#pragma unroll
  for (int ti = 0; ti < 2; ti++)
#pragma unroll
    for (int tj = 0; tj < 2; tj++) {
      v8f acc = (ti == 0) ? (tj == 0 ? acc00 : acc01) : (tj == 0 ? acc10 : acc11);
#pragma unroll
      for (int r = 0; r < 8; r++) {
        long m = m0 + wm0 + 16 * ti + r + 8 * g;
        long n = wn0 + 16 * tj + nl;
        if (m < S_) {
          float v = acc[r];
          if (mode == 0)      qb[((long)bh * S_ + m) * E_ + n] = f2bf_u16(v * scale);
          else if (mode == 1) kb[((long)bh * S_ + m) * E_ + n] = f2bf_u16(v);
          else                vT[((long)bh * E_ + n) * SP_ + m] = f2bf_u16(v);
        }
      }
    }
}

// ---------------------------------------------------------------------------
// GEMM2: scores[s,t] = q'[s,:] . k[t,:]  -> f32 into atten slot of d_out
// ---------------------------------------------------------------------------
__global__ __launch_bounds__(256) void gemm_scores_kernel(
    const u16* __restrict__ qb, const u16* __restrict__ kb,
    float* __restrict__ attn) {
  __shared__ alignas(16) u16 As[64 * 32];
  __shared__ alignas(16) u16 Bs[128 * 32];
  int tid = threadIdx.x, lane = tid & 31, wave = tid >> 5;
  int wm0 = (wave >> 2) * 32, wn0 = (wave & 3) * 32;
  int bh = blockIdx.z;
  long m0 = (long)blockIdx.y * 64;
  long n0 = (long)blockIdx.x * 128;
  const u16* A = qb + (long)bh * S_ * E_;
  const u16* Bt = kb + (long)bh * S_ * E_;   // Bt[n=t][k=f] = k[t][f] (natural)

  v8f acc00 = zero_v8f(), acc01 = zero_v8f(), acc10 = zero_v8f(), acc11 = zero_v8f();
  for (int k0 = 0; k0 < E_; k0 += 32) {
    if (k0 + 32 < E_) {
      __builtin_prefetch(A + m0 * E_ + k0 + 32, 0, 0);
      __builtin_prefetch(Bt + n0 * E_ + k0 + 32, 0, 0);
    }
    load_tile_u16(As, A, E_, m0, k0, S_, E_, 64, tid);
    load_tile_u16(Bs, Bt, E_, n0, k0, S_, E_, 128, tid);
    __syncthreads();
    v16bf a0 = frag_A(As + wm0 * 32, lane);
    v16bf a1 = frag_A(As + (wm0 + 16) * 32, lane);
    v16bf b0 = frag_B(Bs + wn0 * 32, lane);
    v16bf b1 = frag_B(Bs + (wn0 + 16) * 32, lane);
    acc00 = wmma_bf16(a0, b0, acc00);
    acc01 = wmma_bf16(a0, b1, acc01);
    acc10 = wmma_bf16(a1, b0, acc10);
    acc11 = wmma_bf16(a1, b1, acc11);
    __syncthreads();
  }
  int g = lane >> 4, nl = lane & 15;
  float* C = attn + (long)bh * S_ * S_;
#pragma unroll
  for (int ti = 0; ti < 2; ti++)
#pragma unroll
    for (int tj = 0; tj < 2; tj++) {
      v8f acc = (ti == 0) ? (tj == 0 ? acc00 : acc01) : (tj == 0 ? acc10 : acc11);
#pragma unroll
      for (int r = 0; r < 8; r++) {
        long m = m0 + wm0 + 16 * ti + r + 8 * g;
        long n = n0 + wn0 + 16 * tj + nl;
        if (m < S_ && n < S_) C[m * S_ + n] = acc[r];
      }
    }
}

// ---------------------------------------------------------------------------
// Row softmax in place over the atten region: R = B*H*S rows of length S.
// ---------------------------------------------------------------------------
__global__ __launch_bounds__(256) void softmax_rows_kernel(float* __restrict__ attn) {
  __shared__ float red[256];
  long row = blockIdx.x;
  float* p = attn + row * (long)S_;
  int tid = threadIdx.x;
  float v[5];
  float mx = -3.4e38f;
#pragma unroll
  for (int i = 0; i < 5; i++) {
    int idx = tid + i * 256;
    v[i] = (idx < S_) ? p[idx] : -3.4e38f;
    mx = fmaxf(mx, v[i]);
  }
  red[tid] = mx;
  __syncthreads();
  for (int s = 128; s > 0; s >>= 1) {
    if (tid < s) red[tid] = fmaxf(red[tid], red[tid + s]);
    __syncthreads();
  }
  mx = red[0];
  __syncthreads();
  float sum = 0.f;
#pragma unroll
  for (int i = 0; i < 5; i++) {
    int idx = tid + i * 256;
    if (idx < S_) { v[i] = __expf(v[i] - mx); sum += v[i]; }
  }
  red[tid] = sum;
  __syncthreads();
  for (int s = 128; s > 0; s >>= 1) {
    if (tid < s) red[tid] += red[tid + s];
    __syncthreads();
  }
  float inv = 1.f / red[0];
#pragma unroll
  for (int i = 0; i < 5; i++) {
    int idx = tid + i * 256;
    if (idx < S_) p[idx] = v[i] * inv;
  }
}

// ---------------------------------------------------------------------------
// GEMM3: out[s,f] = sum_t atten[s,t] * v[t,f]; A loaded f32->bf16, B = vT rows.
// Output: outc[b][s][h*E + f] bf16.
// ---------------------------------------------------------------------------
__global__ __launch_bounds__(256) void gemm_av_kernel(
    const float* __restrict__ attn, const u16* __restrict__ vT,
    u16* __restrict__ outc) {
  __shared__ alignas(16) u16 As[64 * 32];
  __shared__ alignas(16) u16 Bs[128 * 32];
  int tid = threadIdx.x, lane = tid & 31, wave = tid >> 5;
  int wm0 = (wave >> 2) * 32, wn0 = (wave & 3) * 32;
  int bh = blockIdx.z;
  int h = bh % H_, b = bh / H_;
  long m0 = (long)blockIdx.y * 64;
  const float* A = attn + (long)bh * S_ * S_;
  const u16* Bt = vT + (long)bh * E_ * SP_;   // Bt[n=f][k=t]

  v8f acc00 = zero_v8f(), acc01 = zero_v8f(), acc10 = zero_v8f(), acc11 = zero_v8f();
  for (int k0 = 0; k0 < SP_; k0 += 32) {      // 33 K-steps, zero-padded tail
    load_tile_f32_bf16(As, A, S_, m0, k0, S_, S_, tid);
    load_tile_u16(Bs, Bt, SP_, 0, k0, E_, S_, 128, tid);
    __syncthreads();
    v16bf a0 = frag_A(As + wm0 * 32, lane);
    v16bf a1 = frag_A(As + (wm0 + 16) * 32, lane);
    v16bf b0 = frag_B(Bs + wn0 * 32, lane);
    v16bf b1 = frag_B(Bs + (wn0 + 16) * 32, lane);
    acc00 = wmma_bf16(a0, b0, acc00);
    acc01 = wmma_bf16(a0, b1, acc01);
    acc10 = wmma_bf16(a1, b0, acc10);
    acc11 = wmma_bf16(a1, b1, acc11);
    __syncthreads();
  }
  int g = lane >> 4, nl = lane & 15;
#pragma unroll
  for (int ti = 0; ti < 2; ti++)
#pragma unroll
    for (int tj = 0; tj < 2; tj++) {
      v8f acc = (ti == 0) ? (tj == 0 ? acc00 : acc01) : (tj == 0 ? acc10 : acc11);
#pragma unroll
      for (int r = 0; r < 8; r++) {
        long m = m0 + wm0 + 16 * ti + r + 8 * g;
        long n = wn0 + 16 * tj + nl;
        if (m < S_)
          outc[((long)b * S_ + m) * HE_ + (long)h * E_ + n] = f2bf_u16(acc[r]);
      }
    }
}

// ---------------------------------------------------------------------------
// GEMM4: y[b][e][s] = sum_k outc[b][s][k] * W[e][k] + bias[e] (stored transposed)
// ---------------------------------------------------------------------------
__global__ __launch_bounds__(256) void gemm_final_kernel(
    const u16* __restrict__ outc, const u16* __restrict__ Wb,
    const float* __restrict__ bias, float* __restrict__ y) {
  __shared__ alignas(16) u16 As[64 * 32];
  __shared__ alignas(16) u16 Bs[128 * 32];
  int tid = threadIdx.x, lane = tid & 31, wave = tid >> 5;
  int wm0 = (wave >> 2) * 32, wn0 = (wave & 3) * 32;
  int b = blockIdx.z;
  long m0 = (long)blockIdx.y * 64;
  const u16* A = outc + (long)b * S_ * HE_;
  const u16* Bt = Wb;                         // Bt[n=e][k] = W[e][k] (natural)

  v8f acc00 = zero_v8f(), acc01 = zero_v8f(), acc10 = zero_v8f(), acc11 = zero_v8f();
  for (int k0 = 0; k0 < HE_; k0 += 32) {      // 36 K-steps
    load_tile_u16(As, A, HE_, m0, k0, S_, HE_, 64, tid);
    load_tile_u16(Bs, Bt, HE_, 0, k0, E_, HE_, 128, tid);
    __syncthreads();
    v16bf a0 = frag_A(As + wm0 * 32, lane);
    v16bf a1 = frag_A(As + (wm0 + 16) * 32, lane);
    v16bf b0 = frag_B(Bs + wn0 * 32, lane);
    v16bf b1 = frag_B(Bs + (wn0 + 16) * 32, lane);
    acc00 = wmma_bf16(a0, b0, acc00);
    acc01 = wmma_bf16(a0, b1, acc01);
    acc10 = wmma_bf16(a1, b0, acc10);
    acc11 = wmma_bf16(a1, b1, acc11);
    __syncthreads();
  }
  int g = lane >> 4, nl = lane & 15;
#pragma unroll
  for (int ti = 0; ti < 2; ti++)
#pragma unroll
    for (int tj = 0; tj < 2; tj++) {
      v8f acc = (ti == 0) ? (tj == 0 ? acc00 : acc01) : (tj == 0 ? acc10 : acc11);
#pragma unroll
      for (int r = 0; r < 8; r++) {
        long m = m0 + wm0 + 16 * ti + r + 8 * g;     // s index
        long n = wn0 + 16 * tj + nl;                 // e index
        if (m < S_)
          y[((long)b * E_ + n) * S_ + m] = acc[r] + bias[n];
      }
    }
}

// ---------------------------------------------------------------------------
// Host launch
// ---------------------------------------------------------------------------
extern "C" void kernel_launch(void* const* d_in, const int* in_sizes, int n_in,
                              void* d_out, int out_size, void* d_ws, size_t ws_size,
                              hipStream_t stream) {
  const float* x    = (const float*)d_in[0];   // [B,E,S]
  const float* Q    = (const float*)d_in[1];   // [H,E,E]
  const float* K    = (const float*)d_in[2];   // [H,E,E]
  const float* V    = (const float*)d_in[3];   // [H,E,E]
  const float* W    = (const float*)d_in[4];   // [E,H*E]
  const float* bias = (const float*)d_in[5];   // [E]

  float* out  = (float*)d_out;                           // [B,E,S]
  float* attn = out + (size_t)B_ * E_ * S_;              // [B,H,S,S]

  // Workspace layout (u16/bf16 elements), ~79.4 MB total
  u16* ws = (u16*)d_ws;
  size_t o = 0;
  u16* xt = ws + o;  o += (size_t)B_ * S_ * E_;          // [B,S,E]
  u16* QT = ws + o;  o += (size_t)H_ * E_ * E_;          // [H,F,E] (transposed)
  u16* KT = ws + o;  o += (size_t)H_ * E_ * E_;
  u16* VT = ws + o;  o += (size_t)H_ * E_ * E_;
  u16* qb = ws + o;  o += (size_t)B_ * H_ * S_ * E_;     // scaled q, bf16
  u16* kb = ws + o;  o += (size_t)B_ * H_ * S_ * E_;
  u16* vT = ws + o;  o += (size_t)B_ * H_ * E_ * SP_;    // v transposed, padded stride
  u16* oc = ws + o;  o += (size_t)B_ * S_ * HE_;         // concat head outputs
  u16* Wb = ws + o;  o += (size_t)E_ * HE_;

  // 1) conversions
  {
    long n = (long)B_ * S_ * E_;
    cvt_x_kernel<<<dim3((unsigned)((n + 255) / 256)), 256, 0, stream>>>(x, xt);
  }
  {
    long n = (long)H_ * E_ * E_;
    unsigned g = (unsigned)((n + 255) / 256);
    cvt_headT_kernel<<<g, 256, 0, stream>>>(Q, QT);
    cvt_headT_kernel<<<g, 256, 0, stream>>>(K, KT);
    cvt_headT_kernel<<<g, 256, 0, stream>>>(V, VT);
  }
  {
    long n = (long)E_ * HE_;
    cvt_plain_kernel<<<dim3((unsigned)((n + 255) / 256)), 256, 0, stream>>>(W, Wb, n);
  }

  const unsigned MT = (S_ + 63) / 64;   // 17 row tiles

  // 2) projections: q, k, vT
  gemm_proj_kernel<<<dim3(1, MT, B_ * H_ * 3), 256, 0, stream>>>(xt, QT, KT, VT, qb, kb, vT);

  // 3) scores -> d_out atten region (pre-softmax)
  gemm_scores_kernel<<<dim3((S_ + 127) / 128, MT, B_ * H_), 256, 0, stream>>>(qb, kb, attn);

  // 4) softmax in place
  softmax_rows_kernel<<<dim3((unsigned)((long)B_ * H_ * S_)), 256, 0, stream>>>(attn);

  // 5) atten @ v -> concat head outputs
  gemm_av_kernel<<<dim3(1, MT, B_ * H_), 256, 0, stream>>>(attn, vT, oc);

  // 6) final projection + bias, transposed store
  gemm_final_kernel<<<dim3(1, MT, B_), 256, 0, stream>>>(oc, Wb, bias, out);
}
